// DeepGRU_54932631716303
// MI455X (gfx1250) — matrix operand order repaired and checked by reading, not compiled, not run on hardware
//
#include <hip/hip_runtime.h>
#include <hip/hip_bf16.h>

#define HID    1024
#define INP    512
#define BATCH  128
#define TSTEPS 512

typedef __attribute__((ext_vector_type(16))) _Float16 v16h;
typedef __attribute__((ext_vector_type(8)))  float    v8f;

__device__ __forceinline__ float sig_f(float x) { return 1.0f / (1.0f + __expf(-x)); }

// ===========================================================================
// Big tiled WMMA GEMM for the parallel x-path precompute (M = 65536 rows).
// Block = 256 threads (8 waves). Wave tile = 32x64 (2x4 WMMA). Block = 128x128.
// EPI: 0 = relu -> f16 out, 1 = linear -> f32 out.
// ===========================================================================
template <int EPI>
__global__ __launch_bounds__(256)
void gemm_f16_wmma(const _Float16* __restrict__ A,
                   const _Float16* __restrict__ Wt,
                   const float*    __restrict__ bias,
                   int M, int N, int K,
                   float*          __restrict__ outF,
                   _Float16*       __restrict__ outH)
{
  const int lane = threadIdx.x & 31;
  const int wave = threadIdx.x >> 5;
  const int l15  = lane & 15;
  const int hi   = lane >> 4;

  const int bm = blockIdx.y * 128 + (wave & 3) * 32;
  const int bn = blockIdx.x * 128 + (wave >> 2) * 64;

  v8f acc[2][4];
  const v8f vzero = {0.f,0.f,0.f,0.f,0.f,0.f,0.f,0.f};
  #pragma unroll
  for (int i = 0; i < 2; ++i)
    #pragma unroll
    for (int j = 0; j < 4; ++j) acc[i][j] = vzero;

  const size_t rowA0 = (size_t)(bm +      l15) * K;
  const size_t rowA1 = (size_t)(bm + 16 + l15) * K;
  const int    kbA   = hi * 16;
  const size_t rowB0 = (size_t)(bn +      l15) * K;
  const size_t rowB1 = (size_t)(bn + 16 + l15) * K;
  const size_t rowB2 = (size_t)(bn + 32 + l15) * K;
  const size_t rowB3 = (size_t)(bn + 48 + l15) * K;
  const int    kbB   = hi * 16;

  for (int k0 = 0; k0 < K; k0 += 32) {
    v16h af[2], bf[4];
    {
      const uint4* p0 = (const uint4*)(A + rowA0 + k0 + kbA);
      uint4* d0 = (uint4*)&af[0]; d0[0] = p0[0]; d0[1] = p0[1];
      const uint4* p1 = (const uint4*)(A + rowA1 + k0 + kbA);
      uint4* d1 = (uint4*)&af[1]; d1[0] = p1[0]; d1[1] = p1[1];
    }
    {
      const uint4* q0 = (const uint4*)(Wt + rowB0 + k0 + kbB);
      const uint4* q1 = (const uint4*)(Wt + rowB1 + k0 + kbB);
      const uint4* q2 = (const uint4*)(Wt + rowB2 + k0 + kbB);
      const uint4* q3 = (const uint4*)(Wt + rowB3 + k0 + kbB);
      uint4* e;
      e = (uint4*)&bf[0]; e[0] = q0[0]; e[1] = q0[1];
      e = (uint4*)&bf[1]; e[0] = q1[0]; e[1] = q1[1];
      e = (uint4*)&bf[2]; e[0] = q2[0]; e[1] = q2[1];
      e = (uint4*)&bf[3]; e[0] = q3[0]; e[1] = q3[1];
    }
    #pragma unroll
    for (int i = 0; i < 2; ++i)
      #pragma unroll
      for (int j = 0; j < 4; ++j)
        acc[i][j] = __builtin_amdgcn_wmma_f32_16x16x32_f16(
            false, af[i], false, bf[j], (short)0, acc[i][j], false, false);
  }

  #pragma unroll
  for (int i = 0; i < 2; ++i) {
    #pragma unroll
    for (int j = 0; j < 4; ++j) {
      const int tm = bm + i * 16 + hi * 8;
      const int n  = bn + j * 16 + l15;
      const float bn_ = bias[n];
      #pragma unroll
      for (int v = 0; v < 8; ++v) {
        const int m = tm + v;
        const float pre = acc[i][j][v] + bn_;
        const size_t off = (size_t)m * N + n;
        if (EPI == 0) outH[off] = (_Float16)fmaxf(pre, 0.0f);
        else          outF[off] = pre;
      }
    }
  }
}

// ===========================================================================
// Fused GRU step kernel: 32 blocks x 128 threads (4 waves), wave tile 32x32.
// Four dependent GEMM phases separated by device-scope grid barriers.
// K is a RUNTIME argument so the K-loop is not fully unrolled (avoids the
// register-pressure blowup / scratch spills seen with a constant trip count).
// ===========================================================================
__device__ __forceinline__ void grid_sync(unsigned* __restrict__ slot, unsigned nb) {
  __syncthreads();
  if (threadIdx.x == 0) {
    __threadfence();  // release prior global writes to device scope
    __hip_atomic_fetch_add(slot, 1u, __ATOMIC_RELEASE, __HIP_MEMORY_SCOPE_AGENT);
    while (__hip_atomic_load(slot, __ATOMIC_ACQUIRE, __HIP_MEMORY_SCOPE_AGENT) < nb)
      __builtin_amdgcn_s_sleep(2);
    __threadfence();  // acquire: invalidate stale local cache lines
  }
  __syncthreads();
}

// One 128(M) x 32(N) block-tile GEMM over K, A [128,K] f16, Wt [HID,K] f16.
// EPI: 0 relu->f16 | 2 sigmoid(.+gx)->f32 | 3 f16(h*sigmoid(.+gx)) | 4 GRU update
template <int EPI>
__device__ __forceinline__ void tile_gemm(
    const _Float16* __restrict__ A,
    const _Float16* __restrict__ Wt,
    const float*    __restrict__ bias,
    int K, int m0, int n0,
    float* __restrict__ outF, _Float16* __restrict__ outH,
    const float* __restrict__ gx, long gxS,
    const float* __restrict__ hIn, const float* __restrict__ Zbuf,
    float* __restrict__ hOutF, _Float16* __restrict__ hOutH)
{
  const int lane = threadIdx.x & 31;
  const int l15  = lane & 15;
  const int hi   = lane >> 4;

  v8f acc[2][2];
  const v8f vzero = {0.f,0.f,0.f,0.f,0.f,0.f,0.f,0.f};
  acc[0][0] = vzero; acc[0][1] = vzero; acc[1][0] = vzero; acc[1][1] = vzero;

  const size_t rowA0 = (size_t)(m0 +      l15) * K;
  const size_t rowA1 = (size_t)(m0 + 16 + l15) * K;
  const size_t rowB0 = (size_t)(n0 +      l15) * K;
  const size_t rowB1 = (size_t)(n0 + 16 + l15) * K;
  const int kbA = hi * 16;
  const int kbB = hi * 16;

  for (int k0 = 0; k0 < K; k0 += 32) {
    v16h af[2], bf[2];
    const uint4* p0 = (const uint4*)(A + rowA0 + k0 + kbA);
    uint4* d;
    d = (uint4*)&af[0]; d[0] = p0[0]; d[1] = p0[1];
    const uint4* p1 = (const uint4*)(A + rowA1 + k0 + kbA);
    d = (uint4*)&af[1]; d[0] = p1[0]; d[1] = p1[1];
    const uint4* q0 = (const uint4*)(Wt + rowB0 + k0 + kbB);
    d = (uint4*)&bf[0]; d[0] = q0[0]; d[1] = q0[1];
    const uint4* q1 = (const uint4*)(Wt + rowB1 + k0 + kbB);
    d = (uint4*)&bf[1]; d[0] = q1[0]; d[1] = q1[1];
    #pragma unroll
    for (int i = 0; i < 2; ++i)
      #pragma unroll
      for (int j = 0; j < 2; ++j)
        acc[i][j] = __builtin_amdgcn_wmma_f32_16x16x32_f16(
            false, af[i], false, bf[j], (short)0, acc[i][j], false, false);
  }

  #pragma unroll
  for (int i = 0; i < 2; ++i) {
    #pragma unroll
    for (int j = 0; j < 2; ++j) {
      const int tm = m0 + i * 16 + hi * 8;
      const int n  = n0 + j * 16 + l15;
      const float b = bias[n];
      #pragma unroll
      for (int v = 0; v < 8; ++v) {
        const int m = tm + v;
        const float pre = acc[i][j][v] + b;
        const size_t off = (size_t)m * HID + n;
        if (EPI == 0) {
          outH[off] = (_Float16)fmaxf(pre, 0.0f);
        } else if (EPI == 2) {
          outF[off] = sig_f(pre + gx[(size_t)m * gxS + n]);
        } else if (EPI == 3) {
          outH[off] = (_Float16)(hIn[off] * sig_f(pre + gx[(size_t)m * gxS + n]));
        } else { // 4
          const float hh = tanhf(pre + gx[(size_t)m * gxS + n]);
          const float z  = Zbuf[off];
          const float hn = hIn[off] * z + (1.0f - z) * hh;
          hOutF[off] = hn;
          hOutH[off] = (_Float16)hn;
        }
      }
    }
  }
}

__global__ __launch_bounds__(128)
void gru_step(const _Float16* __restrict__ hH, float* __restrict__ hF,
              const _Float16* __restrict__ Zh1t, const float* __restrict__ Zhb1,
              const _Float16* __restrict__ Zh2t, const float* __restrict__ Zhb2,
              const _Float16* __restrict__ Rh1t, const float* __restrict__ Rhb1,
              const _Float16* __restrict__ Rh2t, const float* __restrict__ Rhb2,
              const _Float16* __restrict__ Hh1t, const float* __restrict__ Hhb1,
              const _Float16* __restrict__ Hh2t, const float* __restrict__ Hhb2,
              const float* __restrict__ gz, const float* __restrict__ gr,
              const float* __restrict__ gh, long gxS,
              _Float16* __restrict__ A1z, _Float16* __restrict__ A1r,
              _Float16* __restrict__ Ub, float* __restrict__ Zb,
              float* __restrict__ hOutF, _Float16* __restrict__ hOutH,
              unsigned* __restrict__ sync3, int K)
{
  const int n0 = blockIdx.x * 32;            // 32 blocks cover N = 1024
  const int m0 = (threadIdx.x >> 5) * 32;    // 4 waves cover M = 128
  const unsigned nb = gridDim.x;

  // Phase A: layer-1 of Z and R paths (independent).
  // __syncthreads between them = scheduling fence -> no cross-GEMM interleave.
  tile_gemm<0>(hH, Zh1t, Zhb1, K, m0, n0, nullptr, A1z, nullptr, 0, nullptr, nullptr, nullptr, nullptr);
  __syncthreads();
  tile_gemm<0>(hH, Rh1t, Rhb1, K, m0, n0, nullptr, A1r, nullptr, 0, nullptr, nullptr, nullptr, nullptr);
  grid_sync(sync3 + 0, nb);

  // Phase B: Z = sigmoid(l2 + gz) ; u = h .* sigmoid(l2 + gr)
  tile_gemm<2>(A1z, Zh2t, Zhb2, K, m0, n0, Zb, nullptr, gz, gxS, nullptr, nullptr, nullptr, nullptr);
  __syncthreads();
  tile_gemm<3>(A1r, Rh2t, Rhb2, K, m0, n0, nullptr, Ub, gr, gxS, hF, nullptr, nullptr, nullptr);
  grid_sync(sync3 + 1, nb);

  // Phase C: layer-1 of H path on u (reuse A1z as A1h)
  tile_gemm<0>(Ub, Hh1t, Hhb1, K, m0, n0, nullptr, A1z, nullptr, 0, nullptr, nullptr, nullptr, nullptr);
  grid_sync(sync3 + 2, nb);

  // Phase D: hh = tanh(l2 + gh); h' = h*Z + (1-Z)*hh
  tile_gemm<4>(A1z, Hh2t, Hhb2, K, m0, n0, nullptr, nullptr, gh, gxS, hF, Zb, hOutF, hOutH);
}

// ===========================================================================
// Conversions / init
// ===========================================================================
__global__ void k_w_to_f16t(const float* __restrict__ W, _Float16* __restrict__ Wt,
                            int K, int N) {
  size_t idx = (size_t)blockIdx.x * 256 + threadIdx.x;
  if (idx >= (size_t)K * N) return;
  int n = (int)(idx / K);
  int k = (int)(idx % K);
  Wt[idx] = (_Float16)W[(size_t)k * N + n];
}

__global__ void k_f32_to_f16(const float* __restrict__ in, _Float16* __restrict__ out,
                             size_t n) {
  size_t idx = (size_t)blockIdx.x * 256 + threadIdx.x;
  if (idx < n) out[idx] = (_Float16)in[idx];
}

__global__ void k_init_h(float* __restrict__ hF, _Float16* __restrict__ hH, size_t n) {
  size_t idx = (size_t)blockIdx.x * 256 + threadIdx.x;
  if (idx < n) { hF[idx] = 0.0f; hH[idx] = (_Float16)0.0f; }
}

__global__ void k_zero_u32(unsigned* __restrict__ p, int n) {
  int i = blockIdx.x * 256 + threadIdx.x;
  if (i < n) p[i] = 0u;
}

extern "C" void kernel_launch(void* const* d_in, const int* in_sizes, int n_in,
                              void* d_out, int out_size, void* d_ws, size_t ws_size,
                              hipStream_t stream) {
  (void)in_sizes; (void)n_in; (void)out_size; (void)ws_size;

  const float* x = (const float*)d_in[0];
  const float* ZhW1=(const float*)d_in[1];  const float* Zhb1=(const float*)d_in[2];
  const float* ZhW2=(const float*)d_in[3];  const float* Zhb2=(const float*)d_in[4];
  const float* ZxW1=(const float*)d_in[5];  const float* Zxb1=(const float*)d_in[6];
  const float* ZxW2=(const float*)d_in[7];  const float* Zxb2=(const float*)d_in[8];
  const float* RhW1=(const float*)d_in[9];  const float* Rhb1=(const float*)d_in[10];
  const float* RhW2=(const float*)d_in[11]; const float* Rhb2=(const float*)d_in[12];
  const float* RxW1=(const float*)d_in[13]; const float* Rxb1=(const float*)d_in[14];
  const float* RxW2=(const float*)d_in[15]; const float* Rxb2=(const float*)d_in[16];
  const float* HhW1=(const float*)d_in[17]; const float* Hhb1=(const float*)d_in[18];
  const float* HhW2=(const float*)d_in[19]; const float* Hhb2=(const float*)d_in[20];
  const float* HxW1=(const float*)d_in[21]; const float* Hxb1=(const float*)d_in[22];
  const float* HxW2=(const float*)d_in[23]; const float* Hxb2=(const float*)d_in[24];

  char* wp = (char*)d_ws;
  auto carve = [&](size_t bytes) -> void* {
    void* r = (void*)wp;
    wp += (bytes + 255) & ~(size_t)255;
    return r;
  };
  const size_t MXT = (size_t)BATCH * TSTEPS;
  _Float16* Zh1t = (_Float16*)carve((size_t)HID*HID*2);
  _Float16* Zh2t = (_Float16*)carve((size_t)HID*HID*2);
  _Float16* Zx1t = (_Float16*)carve((size_t)HID*INP*2);
  _Float16* Zx2t = (_Float16*)carve((size_t)HID*HID*2);
  _Float16* Rh1t = (_Float16*)carve((size_t)HID*HID*2);
  _Float16* Rh2t = (_Float16*)carve((size_t)HID*HID*2);
  _Float16* Rx1t = (_Float16*)carve((size_t)HID*INP*2);
  _Float16* Rx2t = (_Float16*)carve((size_t)HID*HID*2);
  _Float16* Hh1t = (_Float16*)carve((size_t)HID*HID*2);
  _Float16* Hh2t = (_Float16*)carve((size_t)HID*HID*2);
  _Float16* Hx1t = (_Float16*)carve((size_t)HID*INP*2);
  _Float16* Hx2t = (_Float16*)carve((size_t)HID*HID*2);
  _Float16* x16  = (_Float16*)carve(MXT * INP * 2);
  _Float16* A1x  = (_Float16*)carve(MXT * HID * 2);
  float*    GxZ  = (float*)   carve(MXT * HID * 4);
  float*    GxR  = (float*)   carve(MXT * HID * 4);
  float*    GxH  = (float*)   carve(MXT * HID * 4);
  float*    hF   = (float*)   carve((size_t)BATCH*HID*4);
  _Float16* hH   = (_Float16*)carve((size_t)BATCH*HID*2);
  float*    Zb   = (float*)   carve((size_t)BATCH*HID*4);
  _Float16* A1z  = (_Float16*)carve((size_t)BATCH*HID*2);
  _Float16* A1r  = (_Float16*)carve((size_t)BATCH*HID*2);
  _Float16* Ub   = (_Float16*)carve((size_t)BATCH*HID*2);
  unsigned* sync = (unsigned*)carve((size_t)TSTEPS*3*4);

  auto wconv = [&](const float* W, _Float16* Wt, int K, int N) {
    size_t tot = (size_t)K * N;
    k_w_to_f16t<<<dim3((unsigned)((tot + 255) / 256)), 256, 0, stream>>>(W, Wt, K, N);
  };
  wconv(ZhW1, Zh1t, HID, HID); wconv(ZhW2, Zh2t, HID, HID);
  wconv(ZxW1, Zx1t, INP, HID); wconv(ZxW2, Zx2t, HID, HID);
  wconv(RhW1, Rh1t, HID, HID); wconv(RhW2, Rh2t, HID, HID);
  wconv(RxW1, Rx1t, INP, HID); wconv(RxW2, Rx2t, HID, HID);
  wconv(HhW1, Hh1t, HID, HID); wconv(HhW2, Hh2t, HID, HID);
  wconv(HxW1, Hx1t, INP, HID); wconv(HxW2, Hx2t, HID, HID);
  {
    size_t nx = MXT * INP;
    k_f32_to_f16<<<dim3((unsigned)((nx + 255) / 256)), 256, 0, stream>>>(x, x16, nx);
    size_t nh = (size_t)BATCH * HID;
    k_init_h<<<dim3((unsigned)((nh + 255) / 256)), 256, 0, stream>>>(hF, hH, nh);
    int ns = TSTEPS * 3;
    k_zero_u32<<<dim3((ns + 255) / 256), 256, 0, stream>>>(sync, ns);
  }

  const dim3 blk256(256);
  const dim3 gridPre(HID / 128, (unsigned)(MXT / 128));   // (8, 512)
  const long gxS = (long)TSTEPS * HID;

  // ---- x-path precompute (massively parallel) ----
  gemm_f16_wmma<0><<<gridPre, blk256, 0, stream>>>(x16, Zx1t, Zxb1, (int)MXT, HID, INP, nullptr, A1x);
  gemm_f16_wmma<1><<<gridPre, blk256, 0, stream>>>(A1x, Zx2t, Zxb2, (int)MXT, HID, HID, GxZ, nullptr);
  gemm_f16_wmma<0><<<gridPre, blk256, 0, stream>>>(x16, Rx1t, Rxb1, (int)MXT, HID, INP, nullptr, A1x);
  gemm_f16_wmma<1><<<gridPre, blk256, 0, stream>>>(A1x, Rx2t, Rxb2, (int)MXT, HID, HID, GxR, nullptr);
  gemm_f16_wmma<0><<<gridPre, blk256, 0, stream>>>(x16, Hx1t, Hxb1, (int)MXT, HID, INP, nullptr, A1x);
  gemm_f16_wmma<1><<<gridPre, blk256, 0, stream>>>(A1x, Hx2t, Hxb2, (int)MXT, HID, HID, GxH, nullptr);

  // ---- sequential GRU scan: one fused kernel per step ----
  for (int t = 0; t < TSTEPS; ++t) {
    float* hDst = (t == TSTEPS - 1) ? (float*)d_out : hF;
    gru_step<<<dim3(32), dim3(128), 0, stream>>>(
        hH, hF,
        Zh1t, Zhb1, Zh2t, Zhb2,
        Rh1t, Rhb1, Rh2t, Rhb2,
        Hh1t, Hhb1, Hh2t, Hhb2,
        GxZ + (size_t)t * HID, GxR + (size_t)t * HID, GxH + (size_t)t * HID, gxS,
        A1z, A1r, Ub, Zb,
        hDst, hH,
        sync + (size_t)t * 3, HID);
  }
}
